// SAModule_59339268161536
// MI455X (gfx1250) — matrix operand order repaired and checked by reading, not compiled, not run on hardware
//
#include <hip/hip_runtime.h>
#include <hip/hip_bf16.h>

typedef __attribute__((ext_vector_type(16))) __bf16 v16bf;
typedef __attribute__((ext_vector_type(8)))  __bf16 v8bf;
typedef __attribute__((ext_vector_type(8)))  float  v8f;
typedef unsigned long long u64;

#define kN    16384
#define kM    4096
#define kK    64
#define kNIN  64
#define kDIM  67
#define kNOUT 128
#define kR2   0.25f
#define kINVS 0.9999950000374997f   // 1/sqrt(1+1e-5)
#define CAP   4096

// d_out layout (floats/ints, concatenated in reference return order)
#define POSS_OFF  (kM * kNOUT)
#define BATCH_OFF (POSS_OFF + kM * 3)
#define IDX_OFF   (BATCH_OFF + kM)

#define WMMA_BF16(A, B, C) \
  __builtin_amdgcn_wmma_f32_16x16x32_bf16(false, (A), false, (B), (short)0, (C), false, false)

// ---- B-operand swizzle: element (k,n) -> flat offset in (ntile,kc,lane,16) blocks.
// Contiguous-k 16-bit B layout used here, matching load_a96 (KxN=32x16 per wmma): lane = (n&15) + 16*((k%32)/16),
// vgpr = ((k%32)%16)/2, half = k&1.  Lane's 16 bf16 stored contiguously.
__device__ __forceinline__ int bswz(int k, int n) {
  int ntile = n >> 4, kc = k >> 5, kl = k & 31;
  int lane = (n & 15) + ((kl >> 4) << 4);
  int v = (kl & 15) >> 1, h = kl & 1;
  return ((ntile * 3 + kc) * 32 + lane) * 16 + v * 2 + h;
}

// ---- A-operand loader (16-bit A 16x32): per lane 16 contiguous elems:
// K in {2*koff..2*koff+15}, koff = 8*(lane>=16).  Row-major LDS, stride 96.
__device__ __forceinline__ v16bf load_a96(const __bf16* base, int row, int kc, int koff) {
  const __bf16* p = base + row * 96 + kc * 32 + 2 * koff;
  v8bf lo = *(const v8bf*)p;
  v8bf hi = *(const v8bf*)(p + 8);
  return __builtin_shufflevector(lo, hi, 0, 1, 2, 3, 4, 5, 6, 7, 8, 9, 10, 11, 12, 13, 14, 15);
}

__device__ __forceinline__ u64 sx64(u64 v, int m) {
  int lo = __shfl_xor((int)(unsigned)v, m, 32);
  int hi = __shfl_xor((int)(unsigned)(v >> 32), m, 32);
  return ((u64)(unsigned)hi << 32) | (unsigned)lo;
}

// ============================ Kernel 1: prep ============================
// Swizzle W1 (67x67 -> padded 96x80) and W2 (67x128 -> padded 96x128) into the
// WMMA B layout as bf16; build padded (bias, gamma*invs, beta) for layer 1;
// zero the batch output region.
__global__ void prep_kernel(const float* __restrict__ w1, const float* __restrict__ b1,
                            const float* __restrict__ g1, const float* __restrict__ be1,
                            const float* __restrict__ w2,
                            __bf16* __restrict__ W1s, __bf16* __restrict__ W2s,
                            float* __restrict__ b1p, float* __restrict__ g1p,
                            float* __restrict__ be1p, float* __restrict__ batch_out) {
  int t = blockIdx.x * blockDim.x + threadIdx.x;
  const int n1 = 96 * 80, n2 = 96 * 128;
  if (t < n1) {
    int k = t / 80, n = t % 80;
    float v = (k < kDIM && n < kDIM) ? w1[k * kDIM + n] : 0.f;
    W1s[bswz(k, n)] = (__bf16)v;
  } else if (t < n1 + n2) {
    int u = t - n1;
    int k = u / kNOUT, n = u % kNOUT;
    float v = (k < kDIM) ? w2[k * kNOUT + n] : 0.f;
    W2s[bswz(k, n)] = (__bf16)v;
  } else if (t < n1 + n2 + 80) {
    int i = t - n1 - n2;
    b1p[i]  = (i < kDIM) ? b1[i] : 0.f;
    g1p[i]  = (i < kDIM) ? g1[i] * kINVS : 0.f;  // pad scale/shift = 0 so padded H1 cols stay 0
    be1p[i] = (i < kDIM) ? be1[i] : 0.f;
  } else if (t < n1 + n2 + 80 + kM) {
    batch_out[t - n1 - n2 - 80] = 0.f;           // batch is all zeros (int 0 == float 0 bits)
  }
}

// ============================ Kernel 2: FPS =============================
// Single persistent workgroup (32 waves). Each thread owns N/1024 points' dmin
// slots in global scratch (thread-private -> no fences). Per-step argmax via
// packed (f32bits(dmin)<<32 | idx) u64 max reduce: wave shfl_xor + LDS.
__global__ void __launch_bounds__(1024)
fps_kernel(const float* __restrict__ pos, float* __restrict__ dmin,
           int* __restrict__ idx_ws, float* __restrict__ pos_s,
           float* __restrict__ out_poss, int* __restrict__ out_idx) {
  __shared__ u64 red[32];
  __shared__ int s_last;
  const int tid = threadIdx.x;
  for (int i = tid; i < kN; i += 1024) dmin[i] = 3.402823466e+38f;
  int last = 0;
  for (int m = 0; m < kM; ++m) {
    float lx = pos[last * 3 + 0], ly = pos[last * 3 + 1], lz = pos[last * 3 + 2];
    if (tid == 0) {
      idx_ws[m] = last;           out_idx[m] = last;
      pos_s[m * 3 + 0] = lx;      out_poss[m * 3 + 0] = lx;
      pos_s[m * 3 + 1] = ly;      out_poss[m * 3 + 1] = ly;
      pos_s[m * 3 + 2] = lz;      out_poss[m * 3 + 2] = lz;
    }
    u64 best = 0;
    for (int i = tid; i < kN; i += 1024) {
      float dx = pos[i * 3 + 0] - lx;
      float dy = pos[i * 3 + 1] - ly;
      float dz = pos[i * 3 + 2] - lz;
      float d = dx * dx + dy * dy + dz * dz;
      float dm = fminf(dmin[i], d);
      dmin[i] = dm;
      u64 key = ((u64)__float_as_uint(dm) << 32) | (unsigned)i;
      if (key > best) best = key;
    }
    #pragma unroll
    for (int off = 16; off > 0; off >>= 1) {
      u64 o = sx64(best, off);
      if (o > best) best = o;
    }
    if ((tid & 31) == 0) red[tid >> 5] = best;
    __syncthreads();
    if (tid < 32) {
      u64 b = red[tid];
      #pragma unroll
      for (int off = 16; off > 0; off >>= 1) {
        u64 o = sx64(b, off);
        if (o > b) b = o;
      }
      if (tid == 0) s_last = (int)(unsigned)(b & 0xffffffffull);
    }
    __syncthreads();
    last = s_last;
  }
}

// ========================= Kernel 3: ball query =========================
// One workgroup per centroid: collect (d2<=R2) candidates into LDS with an
// atomic counter, bitonic-sort packed (d2bits, idx) u64 keys, emit 64 nearest.
// Missing slots are filled with the centroid's own index (duplicate -> max-safe).
__global__ void __launch_bounds__(256)
ball_kernel(const float* __restrict__ pos, const float* __restrict__ pos_s,
            const int* __restrict__ idx_ws, int* __restrict__ nbr) {
  __shared__ u64 cand[CAP];
  __shared__ int cnt;
  const int m = blockIdx.x, tid = threadIdx.x;
  if (tid == 0) cnt = 0;
  __syncthreads();
  const float cx = pos_s[m * 3 + 0], cy = pos_s[m * 3 + 1], cz = pos_s[m * 3 + 2];
  for (int j = tid; j < kN; j += 256) {
    float dx = pos[j * 3 + 0] - cx;
    float dy = pos[j * 3 + 1] - cy;
    float dz = pos[j * 3 + 2] - cz;
    float d2 = dx * dx + dy * dy + dz * dz;
    if (d2 <= kR2) {
      int p = atomicAdd(&cnt, 1);
      if (p < CAP) cand[p] = ((u64)__float_as_uint(d2) << 32) | (unsigned)j;
    }
  }
  __syncthreads();
  int c = cnt < CAP ? cnt : CAP;
  for (int j = tid; j < CAP; j += 256)
    if (j >= c) cand[j] = ~0ull;
  __syncthreads();
  for (int ksz = 2; ksz <= CAP; ksz <<= 1) {
    for (int j = ksz >> 1; j > 0; j >>= 1) {
      for (int i = tid; i < CAP; i += 256) {
        int ix = i ^ j;
        if (ix > i) {
          bool up = ((i & ksz) == 0);
          u64 a = cand[i], b = cand[ix];
          if (up ? (a > b) : (a < b)) { cand[i] = b; cand[ix] = a; }
        }
      }
      __syncthreads();
    }
  }
  if (tid < kK) {
    u64 key = cand[tid];
    int j = (key == ~0ull) ? idx_ws[m] : (int)(unsigned)(key & 0xffffffffull);
    nbr[m * kK + tid] = j;
  }
}

// ========================= Kernel 4: PointConv ==========================
// One workgroup (4 waves = 4 M-tiles of the 64 neighbors) per centroid.
// GEMM1: [64x96]x[96x80] bf16 WMMA (K padded, cols 67..79 forced to 0 by padded
// scale/shift) -> relu/BN -> bf16 H1 in LDS.  GEMM2: [64x96]x[96x128] -> relu/BN
// -> max over the 64 rows via C-layout per-lane max + shfl_xor(16) + LDS.
__global__ void __launch_bounds__(128)
conv_kernel(const float* __restrict__ x, const float* __restrict__ pos,
            const float* __restrict__ pos_s, const int* __restrict__ nbr,
            const __bf16* __restrict__ W1s, const __bf16* __restrict__ W2s,
            const float* __restrict__ b1p, const float* __restrict__ g1p,
            const float* __restrict__ be1p, const float* __restrict__ b2,
            const float* __restrict__ g2, const float* __restrict__ be2,
            float* __restrict__ xout) {
  __shared__ __bf16 Xa[64 * 96];
  __shared__ __bf16 H1[64 * 96];
  __shared__ float pm[4 * 128];
  const int m = blockIdx.x, tid = threadIdx.x;
  const int lane = tid & 31, wv = tid >> 5;
  const int laneHi = lane >> 4, koff = laneHi * 8;
  const int rowA = wv * 16 + (lane & 15);

  for (int e = tid; e < 64 * 96; e += 128) H1[e] = (__bf16)0.f;
  const float sx0 = pos_s[m * 3 + 0], sy0 = pos_s[m * 3 + 1], sz0 = pos_s[m * 3 + 2];
  for (int e = tid; e < 64 * 96; e += 128) {
    int r = e / 96, c = e % 96;
    int j = nbr[m * kK + r];
    float v = 0.f;
    if (c < kNIN)       v = x[j * kNIN + c];
    else if (c == 64)   v = pos[j * 3 + 0] - sx0;
    else if (c == 65)   v = pos[j * 3 + 1] - sy0;
    else if (c == 66)   v = pos[j * 3 + 2] - sz0;
    Xa[e] = (__bf16)v;
  }
  __syncthreads();

  // ---- GEMM1: Xa[64x96] @ W1s[96x80] ----
  v16bf a0 = load_a96(Xa, rowA, 0, koff);
  v16bf a1 = load_a96(Xa, rowA, 1, koff);
  v16bf a2 = load_a96(Xa, rowA, 2, koff);
  #pragma unroll
  for (int nt = 0; nt < 5; ++nt) {
    v8f acc = {};
    acc = WMMA_BF16(a0, *(const v16bf*)(W1s + ((nt * 3 + 0) * 32 + lane) * 16), acc);
    acc = WMMA_BF16(a1, *(const v16bf*)(W1s + ((nt * 3 + 1) * 32 + lane) * 16), acc);
    acc = WMMA_BF16(a2, *(const v16bf*)(W1s + ((nt * 3 + 2) * 32 + lane) * 16), acc);
    int col = nt * 16 + (lane & 15);
    float bb = b1p[col], gg = g1p[col], be = be1p[col];
    #pragma unroll
    for (int r = 0; r < 8; ++r) {
      int grow = wv * 16 + r + laneHi * 8;
      float v = fmaxf(acc[r] + bb, 0.f);
      H1[grow * 96 + col] = (__bf16)(gg * v + be);
    }
  }
  __syncthreads();

  // ---- GEMM2: H1[64x96] @ W2s[96x128], fused row-max ----
  v16bf h0 = load_a96(H1, rowA, 0, koff);
  v16bf h1 = load_a96(H1, rowA, 1, koff);
  v16bf h2 = load_a96(H1, rowA, 2, koff);
  #pragma unroll
  for (int nt = 0; nt < 8; ++nt) {
    v8f acc = {};
    acc = WMMA_BF16(h0, *(const v16bf*)(W2s + ((nt * 3 + 0) * 32 + lane) * 16), acc);
    acc = WMMA_BF16(h1, *(const v16bf*)(W2s + ((nt * 3 + 1) * 32 + lane) * 16), acc);
    acc = WMMA_BF16(h2, *(const v16bf*)(W2s + ((nt * 3 + 2) * 32 + lane) * 16), acc);
    int col = nt * 16 + (lane & 15);
    float bb = b2[col], gg = g2[col] * kINVS, be = be2[col];
    float pmax = -3.402823466e+38f;
    #pragma unroll
    for (int r = 0; r < 8; ++r) {
      float v = gg * fmaxf(acc[r] + bb, 0.f) + be;
      pmax = fmaxf(pmax, v);
    }
    pmax = fmaxf(pmax, __shfl_xor(pmax, 16, 32));   // combine M=r rows with M=r+8 rows
    if (laneHi == 0) pm[wv * 128 + col] = pmax;
  }
  __syncthreads();
  {
    int c = tid;  // blockDim == 128 == kNOUT
    float v = fmaxf(fmaxf(pm[0 * 128 + c], pm[1 * 128 + c]),
                    fmaxf(pm[2 * 128 + c], pm[3 * 128 + c]));
    xout[(size_t)m * kNOUT + c] = v;
  }
}

// ============================ host launcher =============================
extern "C" void kernel_launch(void* const* d_in, const int* in_sizes, int n_in,
                              void* d_out, int out_size, void* d_ws, size_t ws_size,
                              hipStream_t stream) {
  const float* x   = (const float*)d_in[0];
  const float* pos = (const float*)d_in[1];
  // d_in[2] = batch (all zeros, unused)
  const float* w1  = (const float*)d_in[3];
  const float* b1  = (const float*)d_in[4];
  const float* g1  = (const float*)d_in[5];
  const float* be1 = (const float*)d_in[6];
  const float* w2  = (const float*)d_in[7];
  const float* b2  = (const float*)d_in[8];
  const float* g2  = (const float*)d_in[9];
  const float* be2 = (const float*)d_in[10];

  char* ws = (char*)d_ws;
  int*    idx_ws   = (int*)(ws + 0);               // kM ints          (16 KB)
  float*  dmin_ws  = (float*)(ws + 16384);         // kN floats        (64 KB)
  float*  pos_s_ws = (float*)(ws + 81920);         // kM*3 floats      (48 KB)
  int*    nbr_ws   = (int*)(ws + 131072);          // kM*kK ints       (1 MB)
  __bf16* W1s      = (__bf16*)(ws + 1179648);      // 96*80 bf16       (15 KB)
  __bf16* W2s      = (__bf16*)(ws + 1195008);      // 96*128 bf16      (24 KB)
  float*  b1p      = (float*)(ws + 1219584);       // 80 floats
  float*  g1p      = (float*)(ws + 1219904);       // 80 floats
  float*  be1p     = (float*)(ws + 1220224);       // 80 floats

  float* out_f   = (float*)d_out;
  float* out_x   = out_f;                          // [kM, 128]
  float* out_ps  = out_f + POSS_OFF;               // [kM, 3]
  float* out_bat = out_f + BATCH_OFF;              // [kM] zeros
  int*   out_idx = ((int*)d_out) + IDX_OFF;        // [kM]

  const int prep_total = 96 * 80 + 96 * 128 + 80 + kM;
  prep_kernel<<<(prep_total + 255) / 256, 256, 0, stream>>>(
      w1, b1, g1, be1, w2, W1s, W2s, b1p, g1p, be1p, out_bat);

  fps_kernel<<<1, 1024, 0, stream>>>(pos, dmin_ws, idx_ws, pos_s_ws, out_ps, out_idx);

  ball_kernel<<<kM, 256, 0, stream>>>(pos, pos_s_ws, idx_ws, nbr_ws);

  conv_kernel<<<kM, 128, 0, stream>>>(x, pos, pos_s_ws, nbr_ws, W1s, W2s,
                                      b1p, g1p, be1p, b2, g2, be2, out_x);
}
